// LogOptimalTransport_38465727103433
// MI455X (gfx1250) — compile-verified, matrix operand order checked by reading, not run on hardware
//
#include <hip/hip_runtime.h>

typedef __attribute__((ext_vector_type(16))) _Float16 v16h;
typedef __attribute__((ext_vector_type(8)))  float    v8f;

#define NUM_ITER 100

// ---------------------------------------------------------------------------
// Lane-pair sum 16*(p_n + p_{n+16}), n = lane % 16, via V_WMMA_F32_16X16X32_F16.
// A = all-ones (16x32 f16). B (32x16 f16): lane l holds 16 k-slots of column l%16 of B,
// so filling every slot of lane l with p_l makes column n sum to
// 16*(p_n + p_{n+16}); lane l returns the sum of column l%16. f32 accuracy preserved
// via hi/lo f16 split (two WMMAs, f32 accumulate, ~2^-22 relative error).
// ---------------------------------------------------------------------------
__device__ __forceinline__ float wave_sum_wmma(float x) {
  _Float16 hi = (_Float16)x;
  _Float16 lo = (_Float16)(x - (float)hi);
  v16h A, Bhi, Blo;
#pragma unroll
  for (int k = 0; k < 16; ++k) { A[k] = (_Float16)1.0f; Bhi[k] = hi; Blo[k] = lo; }
  v8f c = {};
  c = __builtin_amdgcn_wmma_f32_16x16x32_f16(false, A, false, Bhi, (short)0, c, false, false);
  c = __builtin_amdgcn_wmma_f32_16x16x32_f16(false, A, false, Blo, (short)0, c, false, false);
  return c[0];
}

__device__ __forceinline__ float wave_max32(float m) {
#pragma unroll
  for (int off = 16; off >= 1; off >>= 1)
    m = fmaxf(m, __shfl_xor(m, off, 32));
  return m;
}

// Online (one-exp) logsumexp accumulate: state (m, p) <- element t.
__device__ __forceinline__ void olse(float t, float& m, float& p) {
  const float d  = t - m;
  const bool  gt = t > m;
  const float e  = __expf(gt ? -d : d);
  p = gt ? __fmaf_rn(p, e, 1.0f) : (p + e);
  m = gt ? t : m;
}

// Merge two (m, s) logsumexp partials into (M, S).
__device__ __forceinline__ void lse_merge(float& M, float& S, float m2, float s2) {
  const float Mn = fmaxf(M, m2);
  S = S * __expf(M - Mn) + s2 * __expf(m2 - Mn);
  M = Mn;
}

// ---------------------------------------------------------------------------
// Row pass: u[b,i] = -logsumexp_j({s[b,i,j] + v[b,j]} U {0})
// One wave per row; 8x float4 per lane. v row (4 KB) staged once per block
// into LDS (all 8 rows of a block share the same batch).
// ---------------------------------------------------------------------------
__global__ void __launch_bounds__(256) sinkhorn_row(const float* __restrict__ s,
                                                    const float* __restrict__ v,
                                                    float* __restrict__ u) {
  __shared__ float4 vsh[256];

  const int lane = threadIdx.x & 31;
  const int wave = threadIdx.x >> 5;
  const int row  = blockIdx.x * 8 + wave;   // row = b*1024 + i
  const int b    = row >> 10;               // same for all 8 waves in block

  const float4* __restrict__ s4 = reinterpret_cast<const float4*>(s) + (size_t)row * 256;
  const float4* __restrict__ v4 = reinterpret_cast<const float4*>(v) + (size_t)b * 256;

  vsh[threadIdx.x] = v4[threadIdx.x];       // stage full v row for this batch
  __syncthreads();

  float4 t[8];
  float m = -3.402823466e38f;
#pragma unroll
  for (int k = 0; k < 8; ++k) {
    float4 a = s4[lane + 32 * k];
    float4 c = vsh[lane + 32 * k];
    a.x += c.x; a.y += c.y; a.z += c.z; a.w += c.w;
    t[k] = a;
    m = fmaxf(m, fmaxf(fmaxf(a.x, a.y), fmaxf(a.z, a.w)));
  }
  float M = wave_max32(m);
  M = fmaxf(M, 0.0f);                       // dustbin column element (value 0)

  float p = 0.0f;
#pragma unroll
  for (int k = 0; k < 8; ++k) {
    p += __expf(t[k].x - M) + __expf(t[k].y - M) +
         __expf(t[k].z - M) + __expf(t[k].w - M);
  }
  float S = wave_sum_wmma(p);
  S += __expf(-M);                          // dustbin term exp(0 - M)
  if (lane == 0) u[row] = -(M + __logf(S));
}

// ---------------------------------------------------------------------------
// Column pass: v[b,j] = -logsumexp_i({s[b,i,j] + u[b,i]} U {0})
// Each thread owns 4 adjacent columns (float4 loads, 512B per wave-load) and
// keeps 4 independent online-lse chains. Block = 128 columns x 16 strips of
// 64 rows; LDS merge of the 16 strip partials, float4 store of v.
// ---------------------------------------------------------------------------
__global__ void __launch_bounds__(512) sinkhorn_col(const float* __restrict__ s,
                                                    const float* __restrict__ u,
                                                    float* __restrict__ v) {
  __shared__ float4 sm_m[16][32];
  __shared__ float4 sm_s[16][32];

  const int tx    = threadIdx.x;            // 0..31  (float4 column group)
  const int strip = threadIdx.y;            // 0..15  (64-row strip)
  const int b     = blockIdx.y;
  const int j4    = blockIdx.x * 32 + tx;   // float4 column index (0..255)

  const float4* __restrict__ sb4 =
      reinterpret_cast<const float4*>(s + ((size_t)b << 20)) + j4;
  const float* __restrict__ ub = u + ((size_t)b << 10);

  float4 m = make_float4(-3.402823466e38f, -3.402823466e38f,
                         -3.402823466e38f, -3.402823466e38f);
  float4 p = make_float4(0.0f, 0.0f, 0.0f, 0.0f);

  const int i0 = strip * 64;
#pragma unroll 4
  for (int k = 0; k < 64; ++k) {
    const int   i  = i0 + k;
    const float ui = ub[i];                 // wave-uniform broadcast load
    const float4 a = sb4[(size_t)i * 256];
    olse(a.x + ui, m.x, p.x);
    olse(a.y + ui, m.y, p.y);
    olse(a.z + ui, m.z, p.z);
    olse(a.w + ui, m.w, p.w);
  }
  sm_m[strip][tx] = m;
  sm_s[strip][tx] = p;
  __syncthreads();

  if (strip == 0) {
#pragma unroll
    for (int st = 1; st < 16; ++st) {
      const float4 m2 = sm_m[st][tx];
      const float4 s2 = sm_s[st][tx];
      lse_merge(m.x, p.x, m2.x, s2.x);
      lse_merge(m.y, p.y, m2.y, s2.y);
      lse_merge(m.z, p.z, m2.z, s2.z);
      lse_merge(m.w, p.w, m2.w, s2.w);
    }
    // dustbin row element (value 0), then v = -(M + log(S))
    float4 o;
    {
      const float Mn = fmaxf(m.x, 0.0f);
      o.x = -(Mn + __logf(p.x * __expf(m.x - Mn) + __expf(-Mn)));
    }
    {
      const float Mn = fmaxf(m.y, 0.0f);
      o.y = -(Mn + __logf(p.y * __expf(m.y - Mn) + __expf(-Mn)));
    }
    {
      const float Mn = fmaxf(m.z, 0.0f);
      o.z = -(Mn + __logf(p.z * __expf(m.z - Mn) + __expf(-Mn)));
    }
    {
      const float Mn = fmaxf(m.w, 0.0f);
      o.w = -(Mn + __logf(p.w * __expf(m.w - Mn) + __expf(-Mn)));
    }
    reinterpret_cast<float4*>(v + ((size_t)b << 10))[j4] = o;
  }
}

// out[b,i,j] = scores[b,i,j] + u[b,i] + v[b,j]
__global__ void __launch_bounds__(256) sinkhorn_out(const float* __restrict__ s,
                                                    const float* __restrict__ u,
                                                    const float* __restrict__ v,
                                                    float* __restrict__ out) {
  const size_t g   = (size_t)blockIdx.x * 256 + threadIdx.x;  // float4 index
  const int    row = (int)(g >> 8);          // b*1024 + i
  const int    b   = row >> 10;
  const int    j4  = (int)(g & 255);

  float4 a  = reinterpret_cast<const float4*>(s)[g];
  const float4 vv = reinterpret_cast<const float4*>(v)[(size_t)b * 256 + j4];
  const float  uu = u[row];
  a.x += uu + vv.x; a.y += uu + vv.y; a.z += uu + vv.z; a.w += uu + vv.w;
  reinterpret_cast<float4*>(out)[g] = a;
}

__global__ void zero_vec(float* __restrict__ p, int n) {
  const int i = blockIdx.x * 256 + threadIdx.x;
  if (i < n) p[i] = 0.0f;
}

extern "C" void kernel_launch(void* const* d_in, const int* in_sizes, int n_in,
                              void* d_out, int out_size, void* d_ws, size_t ws_size,
                              hipStream_t stream) {
  const float* scores = (const float*)d_in[0];
  float*       out    = (float*)d_out;

  const int total = in_sizes[0];      // B * 1024 * 1024
  const int B     = total >> 20;

  float* v = (float*)d_ws;            // B*1024 floats
  float* u = v + (size_t)B * 1024;    // B*1024 floats
  const int nv = B * 1024;

  zero_vec<<<(nv + 255) / 256, 256, 0, stream>>>(v, nv);

  const int  rblocks = (B * 1024) / 8;     // wave-per-row, 8 waves/block
  const dim3 cgrid(8, B), cblk(32, 16);    // 8 x 128-column groups per batch

  for (int it = 0; it < NUM_ITER; ++it) {
    sinkhorn_row<<<rblocks, 256, 0, stream>>>(scores, v, u);
    sinkhorn_col<<<cgrid, cblk, 0, stream>>>(scores, u, v);
  }

  const int oblocks = total / 1024;        // (total/4) float4 / 256 threads
  sinkhorn_out<<<oblocks, 256, 0, stream>>>(scores, u, v, out);
}
